// GraphFilter_69243462746354
// MI455X (gfx1250) — matrix operand order repaired and checked by reading, not compiled, not run on hardware
//
#include <hip/hip_runtime.h>
#include <hip/hip_bf16.h>

#define NNODES 100000
#define NEDGES 3200000
#define FDIM   256
#define MTILES (NNODES / 16)     // 6250 16-row tiles
#define KTILES 32                // 1024 / 32

typedef __attribute__((ext_vector_type(16))) __bf16         v16bf;
typedef __attribute__((ext_vector_type(16))) unsigned short v16u;
typedef __attribute__((ext_vector_type(8)))  float          v8f;

// round-to-nearest-even f32 -> bf16 (raw u16); only used in pack passes
__device__ __forceinline__ unsigned short f2bf(float f) {
    unsigned u = __builtin_bit_cast(unsigned, f);
    u += 0x7FFFu + ((u >> 16) & 1u);
    return (unsigned short)(u >> 16);
}

// ---------------------------------------------------------------------------
// Zero-init a hop buffer (harness poisons ws with 0xAA).
// ---------------------------------------------------------------------------
__global__ void gf_zero_kernel(float4* __restrict__ p, long n4) {
    long i = (long)blockIdx.x * blockDim.x + threadIdx.x;
    long stride = (long)gridDim.x * blockDim.x;
    float4 z = make_float4(0.f, 0.f, 0.f, 0.f);
    for (; i < n4; i += stride) p[i] = z;
}

// ---------------------------------------------------------------------------
// Symmetric SpMM hop: zout[d] += zin[s]; zout[s] += zin[d] for every edge.
// 64 threads per edge, float4 per thread: contiguous 1KB row traffic,
// global_atomic_add_f32 scatter.
// ---------------------------------------------------------------------------
__global__ void gf_spmm_kernel(const float* __restrict__ zin,
                               float* __restrict__ zout,
                               const int* __restrict__ esrc,
                               const int* __restrict__ edst) {
    long gid    = (long)blockIdx.x * blockDim.x + threadIdx.x;
    long group  = gid >> 6;                          // 64 threads / edge
    int  f      = (int)(gid & 63) * 4;
    long gcount = ((long)gridDim.x * blockDim.x) >> 6;

    for (long e = group; e < NEDGES; e += gcount) {
        int s = esrc[e];
        int d = edst[e];
        const float4 a = *(const float4*)(zin + (long)s * FDIM + f);
        const float4 b = *(const float4*)(zin + (long)d * FDIM + f);
        float* od = zout + (long)d * FDIM + f;
        float* os = zout + (long)s * FDIM + f;
        atomicAdd(od + 0, a.x); atomicAdd(od + 1, a.y);
        atomicAdd(od + 2, a.z); atomicAdd(od + 3, a.w);
        atomicAdd(os + 0, b.x); atomicAdd(os + 1, b.y);
        atomicAdd(os + 2, b.z); atomicAdd(os + 3, b.w);
    }
}

// ---------------------------------------------------------------------------
// Pre-swizzle W [1024,256] f32 into bf16 B-fragments for v_wmma 16x16x32:
//   frag (kt, nt): 32 lanes x 16 bf16 (32B/lane contiguous).
//   lane L -> column n = nt*16 + (L%16); element i -> k = kt*32 + i + 8*((i>>3) + L/16).
// ---------------------------------------------------------------------------
__global__ void gf_packw_kernel(const float* __restrict__ w,
                                unsigned short* __restrict__ wp) {
    int o = blockIdx.x * blockDim.x + threadIdx.x;       // 262144 elements
    if (o >= 1024 * 256) return;
    int i    = o & 15;
    int lane = (o >> 4) & 31;
    int nt   = (o >> 9) & 15;
    int kt   = o >> 13;
    int k = kt * 32 + i + 8 * ((i >> 3) + (lane >> 4));
    int n = nt * 16 + (lane & 15);
    wp[o] = f2bf(w[k * 256 + n]);
}

// ---------------------------------------------------------------------------
// Convert one hop's f32 features into bf16 A-fragments.
//   frag (tm, kt): 32 lanes x 16 bf16 (32B/lane contiguous).
//   lane L -> row m = tm*16 + (L%16); element i -> K = i + 8*((i>>3) + L/16)
//   within the k-tile, i.e. the two contiguous 8-float runs below.
// One thread per fragment-lane: 64B f32 in, 32B bf16 out.
// ---------------------------------------------------------------------------
__global__ void gf_packa_kernel(const float* __restrict__ src,
                                unsigned short* __restrict__ ap, int hop) {
    int t = blockIdx.x * blockDim.x + threadIdx.x;       // MTILES*8*32 threads
    if (t >= MTILES * 8 * 32) return;
    int lane = t & 31;
    int ktl  = (t >> 5) & 7;                             // k-tile within hop
    int tm   = t >> 8;
    int kt   = hop * 8 + ktl;

    const float* zr = src + ((long)tm * 16 + (lane & 15)) * FDIM
                          + ktl * 32 + (lane >> 4) * 8;
    float4 a0 = *(const float4*)(zr + 0);
    float4 a1 = *(const float4*)(zr + 4);
    float4 a2 = *(const float4*)(zr + 16);
    float4 a3 = *(const float4*)(zr + 20);
    v16u au;
    au[0]  = f2bf(a0.x); au[1]  = f2bf(a0.y); au[2]  = f2bf(a0.z); au[3]  = f2bf(a0.w);
    au[4]  = f2bf(a1.x); au[5]  = f2bf(a1.y); au[6]  = f2bf(a1.z); au[7]  = f2bf(a1.w);
    au[8]  = f2bf(a2.x); au[9]  = f2bf(a2.y); au[10] = f2bf(a2.z); au[11] = f2bf(a2.w);
    au[12] = f2bf(a3.x); au[13] = f2bf(a3.y); au[14] = f2bf(a3.z); au[15] = f2bf(a3.w);
    *(v16u*)(ap + (((long)tm * KTILES + kt) * 32 + lane) * 16) = au;
}

// ---------------------------------------------------------------------------
// out[N,256] = Abf16[N,1024] @ Wpacked. One wave per 32x64 output tile:
// 8 f32 accumulators, 2 A frags x 4 B frags = 8 WMMAs per k-step,
// pure b128 loads + WMMA in the hot loop (no conversion VALU).
// ---------------------------------------------------------------------------
__global__ void __launch_bounds__(128)
gf_gemm_kernel(const unsigned short* __restrict__ ap,
               const unsigned short* __restrict__ wp,
               float* __restrict__ out) {
    const int lane = threadIdx.x & 31;
    const int wave = threadIdx.x >> 5;       // 0..3 -> n-cols [wave*64, +64)
    const int tm2  = blockIdx.x;             // rows [tm2*32, +32)
    const int nt0  = wave * 4;

    v8f c0 = {}, c1 = {}, c2 = {}, c3 = {}, c4 = {}, c5 = {}, c6 = {}, c7 = {};

    const v16bf* aLo = (const v16bf*)ap + ((long)(2 * tm2 + 0) * KTILES) * 32 + lane;
    const v16bf* aHi = (const v16bf*)ap + ((long)(2 * tm2 + 1) * KTILES) * 32 + lane;
    const v16bf* wfrag = (const v16bf*)wp + lane;

    for (int kt = 0; kt < KTILES; ++kt) {
        v16bf alo = aLo[kt * 32];
        v16bf ahi = aHi[kt * 32];
        const v16bf* wt = wfrag + (long)kt * 16 * 32;
        v16bf b0 = wt[(nt0 + 0) * 32];
        v16bf b1 = wt[(nt0 + 1) * 32];
        v16bf b2 = wt[(nt0 + 2) * 32];
        v16bf b3 = wt[(nt0 + 3) * 32];
        c0 = __builtin_amdgcn_wmma_f32_16x16x32_bf16(false, alo, false, b0, (short)0, c0, false, false);
        c1 = __builtin_amdgcn_wmma_f32_16x16x32_bf16(false, ahi, false, b0, (short)0, c1, false, false);
        c2 = __builtin_amdgcn_wmma_f32_16x16x32_bf16(false, alo, false, b1, (short)0, c2, false, false);
        c3 = __builtin_amdgcn_wmma_f32_16x16x32_bf16(false, ahi, false, b1, (short)0, c3, false, false);
        c4 = __builtin_amdgcn_wmma_f32_16x16x32_bf16(false, alo, false, b2, (short)0, c4, false, false);
        c5 = __builtin_amdgcn_wmma_f32_16x16x32_bf16(false, ahi, false, b2, (short)0, c5, false, false);
        c6 = __builtin_amdgcn_wmma_f32_16x16x32_bf16(false, alo, false, b3, (short)0, c6, false, false);
        c7 = __builtin_amdgcn_wmma_f32_16x16x32_bf16(false, ahi, false, b3, (short)0, c7, false, false);
    }

    // C/D layout: lane L -> col = L%16; VGPR r -> row = r + 8*(L/16)
    const int kh   = lane >> 4;
    const int ncol = (lane & 15);
    const int nb   = nt0 * 16;
    for (int r = 0; r < 8; ++r) {
        long mlo = (long)tm2 * 32 + r + 8 * kh;          // first 16-row half
        long mhi = mlo + 16;                             // second 16-row half
        float* olo = out + mlo * FDIM + nb + ncol;
        float* ohi = out + mhi * FDIM + nb + ncol;
        olo[0] = c0[r]; olo[16] = c2[r]; olo[32] = c4[r]; olo[48] = c6[r];
        ohi[0] = c1[r]; ohi[16] = c3[r]; ohi[32] = c5[r]; ohi[48] = c7[r];
    }
}

// ---------------------------------------------------------------------------
extern "C" void kernel_launch(void* const* d_in, const int* in_sizes, int n_in,
                              void* d_out, int out_size, void* d_ws, size_t ws_size,
                              hipStream_t stream) {
    const float* x    = (const float*)d_in[0];           // [N, 256]
    const int*   edge = (const int*)d_in[1];             // [2, E] flat
    const float* w    = (const float*)d_in[2];           // [4, 256, 256]
    float*       out  = (float*)d_out;                   // [N, 256]

    const long znum = (long)NNODES * FDIM;               // floats per hop buffer
    // ws layout (~410.5 MB): zA | zB (f32 ping-pong) | ap (bf16 A) | wp (bf16 W)
    float* zA = (float*)d_ws;
    float* zB = zA + znum;
    unsigned short* ap = (unsigned short*)(zB + znum);   // N*1024 bf16 = 204.8 MB
    unsigned short* wp = ap + (long)NNODES * 1024;       // 1024*256 bf16 = 512 KB

    const int* esrc = edge;
    const int* edst = edge + NEDGES;

    const int packaBlocks = (MTILES * 8 * 32) / 256;     // 6250
    const long zq = znum / 4;

    // pack weights into WMMA B-fragment layout
    gf_packw_kernel<<<(1024 * 256) / 256, 256, 0, stream>>>(w, wp);
    // hop 0: x straight into A-fragments (k-tiles 0..7)
    gf_packa_kernel<<<packaBlocks, 256, 0, stream>>>(x, ap, 0);
    // hop 1
    gf_zero_kernel<<<4096, 256, 0, stream>>>((float4*)zA, zq);
    gf_spmm_kernel<<<65536, 256, 0, stream>>>(x, zA, esrc, edst);
    gf_packa_kernel<<<packaBlocks, 256, 0, stream>>>(zA, ap, 1);
    // hop 2
    gf_zero_kernel<<<4096, 256, 0, stream>>>((float4*)zB, zq);
    gf_spmm_kernel<<<65536, 256, 0, stream>>>(zA, zB, esrc, edst);
    gf_packa_kernel<<<packaBlocks, 256, 0, stream>>>(zB, ap, 2);
    // hop 3 (zA reused)
    gf_zero_kernel<<<4096, 256, 0, stream>>>((float4*)zA, zq);
    gf_spmm_kernel<<<65536, 256, 0, stream>>>(zB, zA, esrc, edst);
    gf_packa_kernel<<<packaBlocks, 256, 0, stream>>>(zA, ap, 3);
    // fused projection via bf16 WMMA
    gf_gemm_kernel<<<NNODES / 32, 128, 0, stream>>>(ap, wp, out);
}
